// Attention_73023033966874
// MI455X (gfx1250) — compile-verified, matrix operand order checked by reading, not run on hardware
//
#include <hip/hip_runtime.h>

typedef __attribute__((ext_vector_type(16))) _Float16 v16h;
typedef __attribute__((ext_vector_type(8)))  _Float16 v8h;
typedef __attribute__((ext_vector_type(8)))  float    v8f;

#define C_DIM 64
#define NTOK  64
#define HEADS 2
#define DHW   262144   // 64*64*64
#define HW    4096     // 64*64

// strides in halfs
#define XN_S 72
#define QK_S 40
#define VT_S 72
#define AT_S 72
#define OU_S 72

__device__ __forceinline__ v8f wmma16(v16h a, v16h b, v8f c) {
  // v_wmma_f32_16x16x32_f16 : D = A(16x32 f16) * B(32x16 f16) + C(16x16 f32)
  return __builtin_amdgcn_wmma_f32_16x16x32_f16(false, a, false, b, (short)0, c, false, false);
}

// Load an A-fragment (16x32, M rows, K contiguous) or a B-fragment (32x16 with B
// stored transposed as [N][K], K contiguous). Both use identical per-lane
// addressing: lanes 0-15 take K chunks {0..7,16..23}, lanes 16-31 {8..15,24..31}.
__device__ __forceinline__ v16h load_frag(const _Float16* __restrict__ base,
                                          int stride, int row0, int col0, int lane) {
  const int r = row0 + (lane & 15);
  const int c = col0 + ((lane & 16) ? 8 : 0);
  const _Float16* p = base + r * stride + c;
  v8h lo = *(const v8h*)(p);
  v8h hi = *(const v8h*)(p + 16);
  return __builtin_shufflevector(lo, hi, 0,1,2,3,4,5,6,7,8,9,10,11,12,13,14,15);
}

// D-fragment (16x16 f32): VGPR r, lane l -> M = r + 8*(l>=16), N = l&15.
__device__ __forceinline__ void store_frag(_Float16* base, int stride, int row0, int col0,
                                           int lane, v8f acc, float mul) {
  const int rb = row0 + ((lane & 16) ? 8 : 0);
  const int c  = col0 + (lane & 15);
  #pragma unroll
  for (int r = 0; r < 8; ++r)
    base[(rb + r) * stride + c] = (_Float16)(acc[r] * mul);
}

__device__ __forceinline__ void store_frag_T(_Float16* base, int stride, int row0, int col0,
                                             int lane, v8f acc) {
  const int rb = row0 + ((lane & 16) ? 8 : 0);
  const int c  = col0 + (lane & 15);
  #pragma unroll
  for (int r = 0; r < 8; ++r)
    base[c * stride + (rb + r)] = (_Float16)acc[r];
}

// Pre-pass: convert qkv_w (192x64) and out_w (64x64) fp32 -> f16 in workspace.
// Both are already laid out as [N][K] (K contiguous) for our B-fragment loads.
__global__ __launch_bounds__(256) void convert_w(const float* __restrict__ qkv_w,
                                                 const float* __restrict__ out_w,
                                                 _Float16* __restrict__ ws) {
  int i = blockIdx.x * 256 + threadIdx.x;
  if (i < 192 * 64) {
    ws[i] = (_Float16)qkv_w[i];
  } else {
    int j = i - 192 * 64;
    if (j < 64 * 64) ws[192 * 64 + j] = (_Float16)out_w[j];
  }
}

__global__ __launch_bounds__(128) void swin_attn_kernel(
    const float* __restrict__ x, const float* __restrict__ gam, const float* __restrict__ bet,
    const float* __restrict__ outb, const float* __restrict__ pse,
    const _Float16* __restrict__ qkvw, const _Float16* __restrict__ outw,
    float* __restrict__ out)
{
  __shared__ __align__(16) char smem[57344];
  float*    xf32 = (float*)smem;                 // [64][65] fp32 staging (dead after LN)
  _Float16* attn = (_Float16*)smem;              // [64][72] alias over xf32
  _Float16* outl = (_Float16*)(smem + 9216);     // [64][72] alias over xf32
  _Float16* xn   = (_Float16*)(smem + 18432);    // [64][72] normalized tokens
  _Float16* qld  = (_Float16*)(smem + 27648);    // [2][64][40] q (scaled)
  _Float16* kld  = (_Float16*)(smem + 37888);    // [2][64][40] k
  _Float16* vld  = (_Float16*)(smem + 48128);    // [2][32][72] v^T (d-major)

  const int tid  = threadIdx.x;
  const int lane = tid & 31;
  const int wv   = tid >> 5;
  const int mt   = wv << 4;          // this wave's 16-token M-tile

  const int widx = blockIdx.x;
  const int bb = widx >> 12;
  const int wd = (widx >> 8) & 15;
  const int wh = (widx >> 4) & 15;
  const int ww = widx & 15;
  const size_t base = (size_t)bb * 64 * DHW
                    + (size_t)(wd * 4) * HW + (size_t)(wh * 4) * 64 + (size_t)(ww * 4);

  // ---- load 4x4x4x64ch window -> LDS fp32 [c][t], float4 per (c,dz,hy) row ----
  #pragma unroll
  for (int i = 0; i < 8; ++i) {
    int r  = tid + i * 128;          // 0..1023
    int c  = r >> 4;
    int dz = (r >> 2) & 3;
    int hy = r & 3;
    float4 v = *(const float4*)(x + base + (size_t)c * DHW + dz * HW + hy * 64);
    int t = dz * 16 + hy * 4;
    float* dst = xf32 + c * 65 + t;
    dst[0] = v.x; dst[1] = v.y; dst[2] = v.z; dst[3] = v.w;
  }
  __syncthreads();

  // ---- channel LayerNorm per token (+gamma,beta,pse) -> f16 [t][c] ----
  if (tid < NTOK) {
    const int t = tid;
    float s = 0.f, s2 = 0.f;
    for (int c = 0; c < C_DIM; ++c) {
      float v = xf32[c * 65 + t];
      s += v; s2 += v * v;
    }
    float mean = s * (1.f / 64.f);
    float var  = s2 * (1.f / 64.f) - mean * mean;
    float rstd = rsqrtf(var + 1e-5f);
    float pt   = pse[t];
    for (int c = 0; c < C_DIM; ++c) {
      float v = (xf32[c * 65 + t] - mean) * rstd * gam[c] + bet[c] + pt;
      xn[t * XN_S + c] = (_Float16)v;
    }
  }
  __syncthreads();

  // ---- QKV projection: M=64 tok, K=64 ch, N=192 (wave does its 16-row tile) ----
  {
    v16h a0 = load_frag(xn, XN_S, mt, 0,  lane);
    v16h a1 = load_frag(xn, XN_S, mt, 32, lane);
    #pragma unroll
    for (int nt = 0; nt < 12; ++nt) {
      v8f acc = {0.f,0.f,0.f,0.f,0.f,0.f,0.f,0.f};
      acc = wmma16(a0, load_frag(qkvw, 64, nt * 16, 0,  lane), acc);
      acc = wmma16(a1, load_frag(qkvw, 64, nt * 16, 32, lane), acc);
      const int sect = nt >> 2;        // 0=q 1=k 2=v
      const int ob   = (nt & 3) << 4;  // 0,16,32,48 within section
      const int h    = ob >> 5;
      const int c0   = ob & 31;
      if (sect == 0)      store_frag(qld + h * 2560, QK_S, mt, c0, lane, acc, 0.17677669529663687f);
      else if (sect == 1) store_frag(kld + h * 2560, QK_S, mt, c0, lane, acc, 1.f);
      else                store_frag_T(vld + h * 2304, VT_S, mt, c0, lane, acc);
    }
  }
  __syncthreads();   // k,v produced by all waves

  // ---- attention per head: sim -> softmax (registers) -> attn*V ----
  #pragma unroll
  for (int h = 0; h < HEADS; ++h) {
    v16h qa = load_frag(qld + h * 2560, QK_S, mt, 0, lane);  // K=32, one fragment
    v8f s[4];
    #pragma unroll
    for (int jt = 0; jt < 4; ++jt) {
      v8f z = {0.f,0.f,0.f,0.f,0.f,0.f,0.f,0.f};
      s[jt] = wmma16(qa, load_frag(kld + h * 2560, QK_S, jt * 16, 0, lane), z);
    }
    // softmax over j: each row lives on one 16-lane half -> xor-shuffle reduce
    #pragma unroll
    for (int r = 0; r < 8; ++r) {
      float m = fmaxf(fmaxf(s[0][r], s[1][r]), fmaxf(s[2][r], s[3][r]));
      m = fmaxf(m, __shfl_xor(m, 1));
      m = fmaxf(m, __shfl_xor(m, 2));
      m = fmaxf(m, __shfl_xor(m, 4));
      m = fmaxf(m, __shfl_xor(m, 8));
      float e0 = __expf(s[0][r] - m);
      float e1 = __expf(s[1][r] - m);
      float e2 = __expf(s[2][r] - m);
      float e3 = __expf(s[3][r] - m);
      float sum = e0 + e1 + e2 + e3;
      sum += __shfl_xor(sum, 1);
      sum += __shfl_xor(sum, 2);
      sum += __shfl_xor(sum, 4);
      sum += __shfl_xor(sum, 8);
      float inv = 1.f / sum;
      s[0][r] = e0 * inv; s[1][r] = e1 * inv; s[2][r] = e2 * inv; s[3][r] = e3 * inv;
    }
    #pragma unroll
    for (int jt = 0; jt < 4; ++jt)
      store_frag(attn, AT_S, mt, jt * 16, lane, s[jt], 1.f);  // wave-local rows

    v16h p0 = load_frag(attn, AT_S, mt, 0,  lane);
    v16h p1 = load_frag(attn, AT_S, mt, 32, lane);
    #pragma unroll
    for (int dt = 0; dt < 2; ++dt) {
      v8f acc = {0.f,0.f,0.f,0.f,0.f,0.f,0.f,0.f};
      acc = wmma16(p0, load_frag(vld + h * 2304, VT_S, dt * 16, 0,  lane), acc);
      acc = wmma16(p1, load_frag(vld + h * 2304, VT_S, dt * 16, 32, lane), acc);
      store_frag(outl, OU_S, mt, h * 32 + dt * 16, lane, acc, 1.f);
    }
  }
  // outl rows for this tile are entirely wave-local: no barrier needed

  // ---- output projection + bias, coalesced float4 stores ----
  {
    v16h f0 = load_frag(outl, OU_S, mt, 0,  lane);
    v16h f1 = load_frag(outl, OU_S, mt, 32, lane);
    #pragma unroll
    for (int ct = 0; ct < 4; ++ct) {
      v8f acc = {0.f,0.f,0.f,0.f,0.f,0.f,0.f,0.f};
      acc = wmma16(f0, load_frag(outw, 64, ct * 16, 0,  lane), acc);
      acc = wmma16(f1, load_frag(outw, 64, ct * 16, 32, lane), acc);
      const int c   = ct * 16 + (lane & 15);
      const float bc = outb[c];
      const int tb  = mt + ((lane & 16) ? 8 : 0);
      #pragma unroll
      for (int gdx = 0; gdx < 2; ++gdx) {
        int t  = tb + gdx * 4;        // t..t+3 share (dz,hy), wx = 0..3
        int dz = t >> 4;
        int hy = (t >> 2) & 3;
        float4 v;
        v.x = acc[gdx * 4 + 0] + bc;
        v.y = acc[gdx * 4 + 1] + bc;
        v.z = acc[gdx * 4 + 2] + bc;
        v.w = acc[gdx * 4 + 3] + bc;
        *(float4*)(out + base + (size_t)c * DHW + dz * HW + hy * 64) = v;
      }
    }
  }
}

extern "C" void kernel_launch(void* const* d_in, const int* in_sizes, int n_in,
                              void* d_out, int out_size, void* d_ws, size_t ws_size,
                              hipStream_t stream) {
  (void)in_sizes; (void)n_in; (void)out_size; (void)ws_size;
  const float* x    = (const float*)d_in[0];
  const float* g    = (const float*)d_in[1];
  const float* b    = (const float*)d_in[2];
  const float* qkvw = (const float*)d_in[3];
  const float* outw = (const float*)d_in[4];
  const float* outb = (const float*)d_in[5];
  const float* pse  = (const float*)d_in[6];
  _Float16* wsh = (_Float16*)d_ws;

  convert_w<<<64, 256, 0, stream>>>(qkvw, outw, wsh);
  // 4 * 16^3 = 16384 windows, one per workgroup (128 threads = 4 waves)
  swin_attn_kernel<<<16384, 128, 0, stream>>>(x, g, b, outb, pse,
                                              wsh, wsh + 192 * 64, (float*)d_out);
}
